// Net_66821101191377
// MI455X (gfx1250) — matrix operand order repaired and checked by reading, not compiled, not run on hardware
//
#include <hip/hip_runtime.h>

#define B_    128
#define N_    64
#define D_    32
#define IN_   3072
#define F_    5088          // (N-1)*D + IN  (reference reduction length)
#define K_    5120          // N*D + IN      (zero-padded self block -> regular GEMM)
#define ND_   2048          // N*D
#define LBL_  10
#define TSTEPS 3

#define KC_   128           // K-chunk staged in LDS per TDM transfer
#define NCH_  (K_ / KC_)    // 40 chunks
#define RS_   (KC_ + 8)     // padded LDS row stride (halfs): 272B -> conflict-free banks

typedef __attribute__((ext_vector_type(16))) __bf16 v16bf;
typedef __attribute__((ext_vector_type(8)))  float  v8f;
typedef __attribute__((ext_vector_type(4)))  unsigned int v4u;
typedef __attribute__((ext_vector_type(4)))  int    v4i;
typedef __attribute__((ext_vector_type(8)))  int    v8i;

union FragU { uint4 q[2]; v16bf v; };

__device__ __forceinline__ unsigned short f32_to_bf16(float f) {
    unsigned int u = __float_as_uint(f);
    unsigned int r = u + 0x7FFFu + ((u >> 16) & 1u);   // round-to-nearest-even
    return (unsigned short)(r >> 16);
}

// ---------------------------------------------------------------------------
// TDM: DMA a 32-row x KC_-col bf16 tile (row stride K_ elements) into LDS.
// Descriptor per CDNA5 ISA ch.8: group0 {count, lds_addr, global_addr, type=2},
// group1 {data_size=2B, pad 4dw/64dw, tensor/tile dims, dim0 stride}.
// Tracked by TENSORcnt; EXEC ignored (issue once from wave 0).
// ---------------------------------------------------------------------------
__device__ __forceinline__ void tdm_load_b_chunk(const unsigned short* gsrc,
                                                 unsigned lds_addr) {
    unsigned long long ga = (unsigned long long)(uintptr_t)gsrc;
    v4u g0;
    g0[0] = 1u;                                             // count=1, user desc
    g0[1] = lds_addr;                                       // LDS byte address
    g0[2] = (unsigned)(ga & 0xFFFFFFFFu);                   // global_addr[31:0]
    g0[3] = (unsigned)((ga >> 32) & 0x1FFFFFFu) | (2u << 30); // addr[56:32] | type=2
    v8i g1;
    g1[0] = (int)((1u << 16)        // data_size = 2B
                | (1u << 20)        // pad_enable
                | (5u << 22)        // pad_interval: 64 DWORDs (=256B row)
                | (3u << 25));      // pad_amount: 4 DWORDs (=16B)
    g1[1] = (int)(((unsigned)KC_ & 0xFFFFu) << 16);         // tensor_dim0 lo16
    g1[2] = (int)((((unsigned)KC_ >> 16) & 0xFFFFu) | (32u << 16)); // dim0 hi | dim1 lo
    g1[3] = (int)(((unsigned)KC_) << 16);                   // dim1 hi(=0) | tile_dim0=KC
    g1[4] = 32;                                             // tile_dim1=32, tile_dim2=0
    g1[5] = K_;                                             // tensor_dim0_stride lo32
    g1[6] = 0;                                              // stride hi | dim1_stride lo
    g1[7] = 0;
    v4i z4 = {0, 0, 0, 0};
#if defined(__clang_major__) && (__clang_major__ >= 23)
    v8i z8 = {0, 0, 0, 0, 0, 0, 0, 0};
    __builtin_amdgcn_tensor_load_to_lds(g0, g1, z4, z4, z8, 0);
#else
    __builtin_amdgcn_tensor_load_to_lds(g0, g1, z4, z4, 0);
#endif
}

// ---------------------------------------------------------------------------
// Build zero-padded, transposed bf16 weights: Wt[n][d][k], k contiguous.
// ---------------------------------------------------------------------------
__global__ void build_wt_kernel(const float* __restrict__ W,
                                unsigned short* __restrict__ Wt) {
    int o = blockIdx.x * blockDim.x + threadIdx.x;      // total = 64*32*5120
    const int total = N_ * D_ * K_;
    if (o >= total) return;
    int k = o % K_;
    int d = (o / K_) % D_;
    int n = o / (K_ * D_);
    float val = 0.f;
    if (k < ND_) {
        int m = k >> 5, r = k & 31;
        if (m != n) {
            int j = (m < n) ? m : m - 1;                // src_idx = delete(arange(N), n)
            val = W[(n * F_ + j * 32 + r) * D_ + d];
        }
    } else {
        val = W[(n * F_ + (N_ - 1) * D_ + (k - ND_)) * D_ + d];
    }
    Wt[o] = f32_to_bf16(val);
}

// ---------------------------------------------------------------------------
// Per step: zfull[b] = bf16(concat(feat[b], x[b])), and per-(b,n) inverse RMS
// via the identity  sum(z^2) = S_feat - sum(feat[b,n,:]^2) + S_x.
// ---------------------------------------------------------------------------
__global__ void prep_kernel(const float* __restrict__ feat, const float* __restrict__ x,
                            unsigned short* __restrict__ zfull, float* __restrict__ rinv) {
    __shared__ float red[256];
    __shared__ float s_fx;
    const int b = blockIdx.x, t = threadIdx.x;
    float acc = 0.f;
    for (int i = t; i < IN_; i += 256) {
        float v = x[b * IN_ + i];
        acc += v * v;
        zfull[b * K_ + ND_ + i] = f32_to_bf16(v);
    }
    for (int i = t; i < ND_; i += 256) {
        float v = feat[b * ND_ + i];
        acc += v * v;
        zfull[b * K_ + i] = f32_to_bf16(v);
    }
    red[t] = acc;
    __syncthreads();
    for (int s = 128; s > 0; s >>= 1) {
        if (t < s) red[t] += red[t + s];
        __syncthreads();
    }
    if (t == 0) s_fx = red[0];
    __syncthreads();
    if (t < N_) {
        float sn = 0.f;
        const float* fp = feat + b * ND_ + t * D_;
        #pragma unroll
        for (int d = 0; d < D_; ++d) { float v = fp[d]; sn += v * v; }
        float mean = (s_fx - sn) / (float)F_;
        rinv[b * N_ + t] = 1.0f / (sqrtf(mean) + 1e-8f);
    }
}

// ---------------------------------------------------------------------------
// One block per neuron n: 4 waves cover all 128 batch rows (wave = 32-row slab).
// B (Wt[n]) is TDM-staged into LDS, double-buffered, shared by all 4 waves.
// Per wave: 2 M-tiles x 2 N-tiles of v_wmma_f32_16x16x32_bf16.
// Epilogue: * rinv[row,n] + bias[n,d], ReLU.
// ---------------------------------------------------------------------------
__global__ __launch_bounds__(128) void gemm_kernel(
        const unsigned short* __restrict__ zfull,
        const unsigned short* __restrict__ Wt,
        const float* __restrict__ rinv,
        const float* __restrict__ bias,
        float* __restrict__ feat_out) {
    __shared__ unsigned short btile[2][32 * RS_];   // 2 x 8.5KB, padded rows

    const int n    = blockIdx.x;
    const int wave = threadIdx.x >> 5;              // batch slab
    const int lane = threadIdx.x & 31;
    const int half = lane >> 4;
    const int l16  = lane & 15;

    const unsigned short* A0   = zfull + (wave * 32 + l16) * K_;
    const unsigned short* A1   = A0 + 16 * K_;
    const unsigned short* Bsrc = Wt + n * D_ * K_;

    v8f c00 = {}, c01 = {}, c10 = {}, c11 = {};
    const int aoff = half * 8;

    // preload chunk 0
    if (wave == 0) {
        tdm_load_b_chunk(Bsrc, (unsigned)(uintptr_t)&btile[0][0]);
        __builtin_amdgcn_s_wait_tensorcnt(0);
    }
    __syncthreads();

    for (int c = 0; c < NCH_; ++c) {
        const int cur = c & 1;
        if (wave == 0 && (c + 1) < NCH_)
            tdm_load_b_chunk(Bsrc + (c + 1) * KC_,
                             (unsigned)(uintptr_t)&btile[cur ^ 1][0]);

        const unsigned short* Lb0 = &btile[cur][l16 * RS_];        // N-tile 0 col
        const unsigned short* Lb1 = &btile[cur][(16 + l16) * RS_]; // N-tile 1 col
        const int k0 = c * KC_;
        #pragma unroll
        for (int ks = 0; ks < KC_; ks += 32) {
            FragU a0, a1, fb0, fb1;
            a0.q[0]  = *(const uint4*)(A0 + k0 + ks + aoff);
            a0.q[1]  = *(const uint4*)(A0 + k0 + ks + 16 + aoff);
            a1.q[0]  = *(const uint4*)(A1 + k0 + ks + aoff);
            a1.q[1]  = *(const uint4*)(A1 + k0 + ks + 16 + aoff);
            const int bo = ks + half * 8;
            fb0.q[0] = *(const uint4*)(Lb0 + bo);
            fb0.q[1] = *(const uint4*)(Lb0 + bo + 16);
            fb1.q[0] = *(const uint4*)(Lb1 + bo);
            fb1.q[1] = *(const uint4*)(Lb1 + bo + 16);
            c00 = __builtin_amdgcn_wmma_f32_16x16x32_bf16(false, a0.v, false, fb0.v, (short)0, c00, false, false);
            c01 = __builtin_amdgcn_wmma_f32_16x16x32_bf16(false, a0.v, false, fb1.v, (short)0, c01, false, false);
            c10 = __builtin_amdgcn_wmma_f32_16x16x32_bf16(false, a1.v, false, fb0.v, (short)0, c10, false, false);
            c11 = __builtin_amdgcn_wmma_f32_16x16x32_bf16(false, a1.v, false, fb1.v, (short)0, c11, false, false);
        }
        if (wave == 0) __builtin_amdgcn_s_wait_tensorcnt(0);  // chunk c+1 landed
        __syncthreads();                                      // publish / allow reuse
    }

    const float bia0 = bias[n * D_ + l16];
    const float bia1 = bias[n * D_ + 16 + l16];
    #pragma unroll
    for (int r = 0; r < 8; ++r) {
        int row0 = wave * 32 + half * 8 + r;   // C/D layout: M = r + 8*half
        int row1 = row0 + 16;
        float s0 = rinv[row0 * N_ + n];
        float s1 = rinv[row1 * N_ + n];
        float v;
        v = c00[r] * s0 + bia0; feat_out[row0 * ND_ + n * D_ + l16]      = v > 0.f ? v : 0.f;
        v = c01[r] * s0 + bia1; feat_out[row0 * ND_ + n * D_ + 16 + l16] = v > 0.f ? v : 0.f;
        v = c10[r] * s1 + bia0; feat_out[row1 * ND_ + n * D_ + l16]      = v > 0.f ? v : 0.f;
        v = c11[r] * s1 + bia1; feat_out[row1 * ND_ + n * D_ + 16 + l16] = v > 0.f ? v : 0.f;
    }
}

// ---------------------------------------------------------------------------
// Readout: logits[b,:] = feat[b,:] @ Wc + bc   (2048 x 10, negligible cost)
// ---------------------------------------------------------------------------
__global__ void cls_kernel(const float* __restrict__ feat, const float* __restrict__ Wc,
                           const float* __restrict__ bc, float* __restrict__ out) {
    __shared__ float red[256];
    const int b = blockIdx.x, t = threadIdx.x;
    float acc[LBL_];
    #pragma unroll
    for (int c = 0; c < LBL_; ++c) acc[c] = 0.f;
    for (int j = t; j < ND_; j += 256) {
        float f = feat[b * ND_ + j];
        #pragma unroll
        for (int c = 0; c < LBL_; ++c) acc[c] += f * Wc[j * LBL_ + c];
    }
    for (int c = 0; c < LBL_; ++c) {
        red[t] = acc[c];
        __syncthreads();
        for (int s = 128; s > 0; s >>= 1) {
            if (t < s) red[t] += red[t + s];
            __syncthreads();
        }
        if (t == 0) out[b * LBL_ + c] = red[0] + bc[c];
        __syncthreads();
    }
}

extern "C" void kernel_launch(void* const* d_in, const int* in_sizes, int n_in,
                              void* d_out, int out_size, void* d_ws, size_t ws_size,
                              hipStream_t stream) {
    (void)in_sizes; (void)n_in; (void)out_size; (void)ws_size;
    const float* x     = (const float*)d_in[0];
    const float* feat0 = (const float*)d_in[1];
    const float* W     = (const float*)d_in[2];
    const float* bias  = (const float*)d_in[3];
    const float* Wc    = (const float*)d_in[4];
    const float* bc    = (const float*)d_in[5];
    // d_in[6] = src_idx (handled analytically), d_in[7] = T (reference constant 3)

    char* ws = (char*)d_ws;
    size_t off = 0;
    unsigned short* Wt = (unsigned short*)(ws + off); off += (size_t)N_ * D_ * K_ * 2;  // 20.97 MB
    unsigned short* zf = (unsigned short*)(ws + off); off += (size_t)B_ * K_ * 2;       //  1.31 MB
    float* rinv = (float*)(ws + off); off += (size_t)B_ * N_ * 4;                       //   32 KB
    float* fA   = (float*)(ws + off); off += (size_t)B_ * ND_ * 4;                      //    1 MB
    float* fB   = (float*)(ws + off); off += (size_t)B_ * ND_ * 4;                      //    1 MB

    {
        int total  = N_ * D_ * K_;
        int blocks = (total + 255) / 256;
        build_wt_kernel<<<blocks, 256, 0, stream>>>(W, Wt);
    }

    const float* fcur = feat0;
    float* fnxt = fA;
    for (int t = 0; t < TSTEPS; ++t) {
        prep_kernel<<<B_, 256, 0, stream>>>(fcur, x, zf, rinv);
        gemm_kernel<<<N_, 128, 0, stream>>>(zf, Wt, rinv, bias, fnxt);
        fcur = fnxt;
        fnxt = (fnxt == fA) ? fB : fA;
    }
    cls_kernel<<<B_, 256, 0, stream>>>(fcur, Wc, bc, (float*)d_out);
}
